// Model_62156766708388
// MI455X (gfx1250) — matrix operand order repaired and checked by reading, not compiled, not run on hardware
//
#include <hip/hip_runtime.h>

// ---------------------------------------------------------------------------
// 2-layer transformer w/ MoE on gfx1250.  B=4 S=1024 D=1024 H=16 L=2 E=8 K=2
// DFF=4096.  All matmuls run on v_wmma_f32_16x16x32_f16 (f16 in, f32 acc).
// Weights are converted fp32->f16 and pre-transposed to [N][K] once per call
// (weight bytes ~0.8GB @23.3TB/s ~= 35us, negligible vs compute), so the GEMM
// inner loop is pure 16B copies global->LDS (async-to-LDS when available,
// ASYNCcnt-tracked), double-buffered so copies for tile k+1 overlap WMMA on
// tile k.  Activations are produced directly in f16 by fused epilogues.
// Seq-MoE layers skip whole expert GEMMs whose combine weight is exactly 0.
// ---------------------------------------------------------------------------

typedef __attribute__((ext_vector_type(16))) _Float16 v16h;
typedef __attribute__((ext_vector_type(8)))  _Float16 h8;
typedef __attribute__((ext_vector_type(8)))  float    v8f;
typedef __attribute__((ext_vector_type(4)))  int      v4i;

#if __has_builtin(__builtin_amdgcn_global_load_async_to_lds_b128)
#define ASYNC_OK 1
#else
#define ASYNC_OK 0
#endif

enum : unsigned {
  F_BIAS     = 1u,
  F_GELU     = 2u,
  F_ACCUM    = 4u,
  F_ROWSCALE = 8u,
  F_CAUSAL   = 16u,
  F_SKIPZERO = 32u,
  F_C16      = 64u,   // write C as f16 (no accumulate)
};

#define BM 128
#define BN 64
#define BK 32
#define LDS_STRIDE 40  // 32 halves + 8 pad (80B rows keep 16B alignment)

// 16-byte global->LDS copy: async-to-LDS if the toolchain exposes it.
// Builtin signature (leaked by the round-2 diagnostic): param0 is an int4*
// in the global (__device__) address space; param1 is the LDS destination.
__device__ __forceinline__ void copy16(const _Float16* g, _Float16* l) {
#if ASYNC_OK
  __builtin_amdgcn_global_load_async_to_lds_b128(
      (__attribute__((address_space(1))) v4i*)(unsigned long long)g,
      (__attribute__((address_space(3))) v4i*)(unsigned long long)l, 0, 0);
#else
  *reinterpret_cast<h8*>(l) = *reinterpret_cast<const h8*>(g);
#endif
}

__device__ __forceinline__ void wait_copies() {
#if ASYNC_OK
  asm volatile("s_wait_asynccnt 0" ::: "memory");
#endif
}

// ---------------------------------------------------------------------------
// Batched, flagged WMMA GEMM:  C = epilogue(A @ B^T-layout)
//   A: f16 [M x Kd] row-major (lda);  B: f16 [N x Kd] row-major (ldb)
//   C: fp32 (or f16 with F_C16), ldc;  batch z -> (z/Hdiv, z%Hdiv) strides.
// All dims used here are multiples of BM/BN/BK -> no edge guards.
// ---------------------------------------------------------------------------
__global__ __launch_bounds__(256)
void wmma_gemm(const _Float16* __restrict__ A, long sA1, long sA2, int lda,
               const _Float16* __restrict__ Bm, long sB1, long sB2, int ldb,
               const float* __restrict__ bias,
               void* __restrict__ Cv, long sC1, long sC2, int ldc,
               int M, int N, int Kd, int Hdiv,
               unsigned flags, float alpha,
               const float* __restrict__ rowscale, int rs_div)
{
  __shared__ _Float16 ldsA[2][BM][LDS_STRIDE];
  __shared__ _Float16 ldsB[2][BN][LDS_STRIDE];

  const int tid  = threadIdx.x;
  const int lane = tid & 31;
  const int wave = tid >> 5;
  const int wm   = wave >> 1;   // 0..3 -> m offset wm*32
  const int wn   = wave & 1;    // 0..1 -> n offset wn*32
  const int m0   = blockIdx.x * BM;
  const int n0   = blockIdx.y * BN;
  const int z    = blockIdx.z;
  const int zb   = z / Hdiv;
  const int zh   = z - zb * Hdiv;

  A  += (long)zb * sA1 + (long)zh * sA2;
  Bm += (long)zb * sB1 + (long)zh * sB2;
  const long cOff = (long)zb * sC1 + (long)zh * sC2;

  // Uniform early-out: whole tile has zero combine weight (seq-MoE).
  if (flags & F_SKIPZERO) {
    if (rowscale[(m0 / rs_div) * 8] == 0.0f) return;
  }

  // Per-thread staging coordinates (16B chunks of 8 halves).
  const int ar0 = (tid + 0)   >> 2, ac0 = ((tid + 0)   & 3) * 8;  // A chunk 0
  const int ar1 = (tid + 256) >> 2, ac1 = ((tid + 256) & 3) * 8;  // A chunk 1
  const int bnr = tid >> 2,         bc  = (tid & 3) * 8;          // B chunk

  const _Float16* aP0 = A  + (long)(m0 + ar0) * lda + ac0;
  const _Float16* aP1 = A  + (long)(m0 + ar1) * lda + ac1;
  const _Float16* bP  = Bm + (long)(n0 + bnr) * ldb + bc;

  // Prologue: stage k-tile 0 into buffer 0.
  copy16(aP0, &ldsA[0][ar0][ac0]);
  copy16(aP1, &ldsA[0][ar1][ac1]);
  copy16(bP,  &ldsB[0][bnr][bc]);
  wait_copies();
  __syncthreads();

  v8f acc[2][2] = {};
  int cur = 0;

  for (int k0 = 0; k0 < Kd; k0 += BK) {
    const bool haveNext = (k0 + BK) < Kd;
    if (haveNext) {  // fire copies for tile k+1 into the other buffer
      copy16(aP0 + k0 + BK, &ldsA[cur ^ 1][ar0][ac0]);
      copy16(aP1 + k0 + BK, &ldsA[cur ^ 1][ar1][ac1]);
      copy16(bP  + k0 + BK, &ldsB[cur ^ 1][bnr][bc]);
    }

    // Fragment loads per ISA 16-bit A(16x32)/B(32x16) VGPR layouts.
    v16h af[2], bf[2];
    {
      const int mrow = lane & 15;
      const int kbA  = (lane >> 4) * 8;
#pragma unroll
      for (int sm = 0; sm < 2; ++sm) {
        const _Float16* p = &ldsA[cur][wm * 32 + sm * 16 + mrow][kbA];
#pragma unroll
        for (int i = 0; i < 8; ++i) { af[sm][i] = p[i]; af[sm][i + 8] = p[i + 16]; }
      }
      const int kbB = (lane >> 4) * 8;
#pragma unroll
      for (int sn = 0; sn < 2; ++sn) {
        const _Float16* p = &ldsB[cur][wn * 32 + sn * 16 + (lane & 15)][kbB];
#pragma unroll
        for (int i = 0; i < 8; ++i) { bf[sn][i] = p[i]; bf[sn][i + 8] = p[i + 16]; }
      }
    }
#pragma unroll
    for (int sm = 0; sm < 2; ++sm)
#pragma unroll
      for (int sn = 0; sn < 2; ++sn)
        acc[sm][sn] = __builtin_amdgcn_wmma_f32_16x16x32_f16(
            false, af[sm], false, bf[sn], (short)0, acc[sm][sn], false, false);

    if (haveNext) wait_copies();
    __syncthreads();
    cur ^= 1;
  }

  // Epilogue: scale / causal / bias / gelu / rowscale / store (f32|f16).
  const int mhi = (lane >> 4) * 8;
  const int nc  = lane & 15;
#pragma unroll
  for (int sm = 0; sm < 2; ++sm) {
#pragma unroll
    for (int sn = 0; sn < 2; ++sn) {
#pragma unroll
      for (int j = 0; j < 8; ++j) {
        int gm = m0 + wm * 32 + sm * 16 + mhi + j;
        int gn = n0 + wn * 32 + sn * 16 + nc;
        float vv = acc[sm][sn][j] * alpha;
        if ((flags & F_CAUSAL) && gn > gm) vv = -__builtin_inff();
        if (flags & F_BIAS) vv += bias[gn];
        if (flags & F_GELU) vv = 0.5f * vv * (1.0f + erff(vv * 0.70710678118654752f));
        if (flags & F_ROWSCALE) vv *= rowscale[(gm / rs_div) * 8];
        const long idx = cOff + (long)gm * ldc + gn;
        if (flags & F_C16)        ((_Float16*)Cv)[idx] = (_Float16)vv;
        else if (flags & F_ACCUM) ((float*)Cv)[idx]   += vv;
        else                      ((float*)Cv)[idx]    = vv;
      }
    }
  }
}

// ---------------------------------------------------------------------------
// Weight transform: fp32 [K][N] row-major -> f16 [N][K] row-major.
// ---------------------------------------------------------------------------
__global__ __launch_bounds__(256)
void transpose_cvt(const float* __restrict__ in, _Float16* __restrict__ out,
                   int K, int N)
{
  __shared__ float tile[32][33];
  const int k0 = blockIdx.x * 32, n0 = blockIdx.y * 32;
  const int tx = threadIdx.x & 31, ty = threadIdx.x >> 5;
  for (int j = ty; j < 32; j += 8)
    tile[j][tx] = in[(long)(k0 + j) * N + n0 + tx];
  __syncthreads();
  for (int j = ty; j < 32; j += 8)
    out[(long)(n0 + j) * K + k0 + tx] = (_Float16)tile[tx][j];
}

// f16 per-head transpose: v[b,s,h*hd+n] -> vT[(b*H+h)*hd + n][s]
__global__ __launch_bounds__(256)
void transpose_head16(const _Float16* __restrict__ v, _Float16* __restrict__ vT,
                      int S, int D, int hd, int H)
{
  __shared__ _Float16 tile[32][33];
  const int z = blockIdx.z, b = z / H, h = z - b * H;
  const _Float16* in = v + (long)b * S * D + h * hd;     // [S][hd], stride D
  _Float16* out = vT + (long)z * hd * S;                 // [hd][S]
  const int s0 = blockIdx.x * 32, n0 = blockIdx.y * 32;
  const int tx = threadIdx.x & 31, ty = threadIdx.x >> 5;
  for (int j = ty; j < 32; j += 8)
    tile[j][tx] = in[(long)(s0 + j) * D + n0 + tx];
  __syncthreads();
  for (int j = ty; j < 32; j += 8)
    out[(long)(n0 + j) * S + s0 + tx] = tile[tx][j];
}

// ---------------------------------------------------------------------------
// LayerNorm: one block per row; writes fp32 (for gating) and f16 (GEMM A).
// ---------------------------------------------------------------------------
__global__ __launch_bounds__(256)
void layernorm_k(const float* __restrict__ x, const float* __restrict__ w,
                 const float* __restrict__ b, float* __restrict__ o32,
                 _Float16* __restrict__ o16, int Dd)
{
  __shared__ float s1[256], s2[256];
  const int tid = threadIdx.x;
  const float* xp = x + (long)blockIdx.x * Dd;
  float a = 0.f, a2 = 0.f;
  for (int d = tid; d < Dd; d += 256) { float v = xp[d]; a += v; a2 += v * v; }
  s1[tid] = a; s2[tid] = a2;
  __syncthreads();
  for (int o = 128; o > 0; o >>= 1) {
    if (tid < o) { s1[tid] += s1[tid + o]; s2[tid] += s2[tid + o]; }
    __syncthreads();
  }
  const float mean = s1[0] / Dd;
  const float var  = s2[0] / Dd - mean * mean;
  const float rstd = rsqrtf(var + 1e-5f);
  float*     p32 = o32 + (long)blockIdx.x * Dd;
  _Float16*  p16 = o16 + (long)blockIdx.x * Dd;
  for (int d = tid; d < Dd; d += 256) {
    const float v = (xp[d] - mean) * rstd * w[d] + b[d];
    p32[d] = v; p16[d] = (_Float16)v;
  }
}

// Row softmax over f16 scores, in place (math in fp32): wave per row.
__global__ __launch_bounds__(256)
void softmax_rows16(_Float16* __restrict__ att, int Sd)
{
  const int lane = threadIdx.x & 31;
  const int wave = threadIdx.x >> 5;
  _Float16* p = att + ((long)blockIdx.x * 8 + wave) * Sd;
  float mx = -__builtin_inff();
  for (int j = lane; j < Sd; j += 32) mx = fmaxf(mx, (float)p[j]);
#pragma unroll
  for (int o = 16; o > 0; o >>= 1) mx = fmaxf(mx, __shfl_xor(mx, o, 32));
  float sum = 0.f;
  for (int j = lane; j < Sd; j += 32) {
    const float e = __expf((float)p[j] - mx);
    p[j] = (_Float16)e; sum += e;
  }
#pragma unroll
  for (int o = 16; o > 0; o >>= 1) sum += __shfl_xor(sum, o, 32);
  const float inv = 1.f / sum;
  for (int j = lane; j < Sd; j += 32) p[j] = (_Float16)((float)p[j] * inv);
}

// Gate logits + top-2 softmax combine weights; one wave per token.
__global__ __launch_bounds__(256)
void gate_top2(const float* __restrict__ h, const float* __restrict__ gw,
               const float* __restrict__ gb, float* __restrict__ cw,
               int ntok, int Dd)
{
  const int lane = threadIdx.x & 31;
  const int wave = threadIdx.x >> 5;
  const int t = blockIdx.x * 8 + wave;
  if (t >= ntok) return;
  const float* hp = h + (long)t * Dd;
  float acc[8] = {};
  for (int d = lane; d < Dd; d += 32) {
    const float hv = hp[d];
    const float* g = gw + (long)d * 8;
#pragma unroll
    for (int e = 0; e < 8; ++e) acc[e] += hv * g[e];
  }
#pragma unroll
  for (int e = 0; e < 8; ++e)
#pragma unroll
    for (int o = 16; o > 0; o >>= 1) acc[e] += __shfl_xor(acc[e], o, 32);
  if (lane == 0) {
    float lg[8];
#pragma unroll
    for (int e = 0; e < 8; ++e) lg[e] = acc[e] + gb[e];
    int i1 = 0;
#pragma unroll
    for (int e = 1; e < 8; ++e) if (lg[e] > lg[i1]) i1 = e;
    int i2 = -1;
#pragma unroll
    for (int e = 0; e < 8; ++e)
      if (e != i1 && (i2 < 0 || lg[e] > lg[i2])) i2 = e;
    const float m  = lg[i1];
    const float e1 = __expf(lg[i1] - m), e2 = __expf(lg[i2] - m);
    const float inv = 1.f / (e1 + e2);
    float* o = cw + (long)t * 8;
#pragma unroll
    for (int e = 0; e < 8; ++e) o[e] = 0.f;
    o[i1] = e1 * inv; o[i2] = e2 * inv;
  }
}

__global__ __launch_bounds__(256)
void seq_mean(const float* __restrict__ h, float* __restrict__ out,
              int Bn, int Sn, int Dn)
{
  const int idx = blockIdx.x * 256 + threadIdx.x;
  if (idx >= Bn * Dn) return;
  const int b = idx / Dn, d = idx - b * Dn;
  const float* p = h + (long)b * Sn * Dn + d;
  float s = 0.f;
  for (int i = 0; i < Sn; ++i) s += p[(long)i * Dn];
  out[idx] = s / (float)Sn;
}

__global__ __launch_bounds__(256)
void copy_k(float* __restrict__ dst, const float* __restrict__ src, long n)
{ long i = (long)blockIdx.x * 256 + threadIdx.x; if (i < n) dst[i] = src[i]; }

__global__ __launch_bounds__(256)
void zero_k(float* __restrict__ dst, long n)
{ long i = (long)blockIdx.x * 256 + threadIdx.x; if (i < n) dst[i] = 0.f; }

__global__ __launch_bounds__(256)
void add_k(float* __restrict__ dst, const float* __restrict__ src, long n)
{ long i = (long)blockIdx.x * 256 + threadIdx.x; if (i < n) dst[i] += src[i]; }

// ---------------------------------------------------------------------------
extern "C" void kernel_launch(void* const* d_in, const int* in_sizes, int n_in,
                              void* d_out, int out_size, void* d_ws, size_t ws_size,
                              hipStream_t stream) {
  (void)in_sizes; (void)n_in; (void)out_size; (void)ws_size;
  const int B = 4, S = 1024, D = 1024, H = 16, L = 2, E = 8, DFF = 4096;
  const int hd = D / H, BS = B * S, BH = B * H;
  const long BSD = (long)BS * D;
  const long LW  = 4L * D * D + (long)E * (2L * D * DFF);  // f16 weights / layer

  const float* x_in  = (const float*)d_in[0];
  const float* ln1_w = (const float*)d_in[1];
  const float* ln1_b = (const float*)d_in[2];
  const float* ln2_w = (const float*)d_in[3];
  const float* ln2_b = (const float*)d_in[4];
  const float* wq    = (const float*)d_in[5];
  const float* wk    = (const float*)d_in[6];
  const float* wv    = (const float*)d_in[7];
  const float* wo    = (const float*)d_in[8];
  const float* bq    = (const float*)d_in[9];
  const float* bk    = (const float*)d_in[10];
  const float* bv    = (const float*)d_in[11];
  const float* bo    = (const float*)d_in[12];
  const float* gate_w = (const float*)d_in[13];
  const float* gate_b = (const float*)d_in[14];
  const float* e_w1  = (const float*)d_in[15];
  const float* e_b1  = (const float*)d_in[16];
  const float* e_w2  = (const float*)d_in[17];
  const float* e_b2  = (const float*)d_in[18];

  float* x = (float*)d_out;                 // residual stream, in place
  float* f = (float*)d_ws;
  float* h32   = f;  f += BSD;
  float* macc  = f;  f += BSD;
  float* hmean = f;  f += (long)B * D;
  float* cw    = f;  f += (long)BS * E;
  float* cw2   = f;  f += (long)B * E;
  _Float16* hp   = (_Float16*)f;
  _Float16* h16  = hp;  hp += BSD;
  _Float16* q16  = hp;  hp += BSD;
  _Float16* k16  = hp;  hp += BSD;
  _Float16* v16  = hp;  hp += BSD;
  _Float16* vT16 = hp;  hp += BSD;
  _Float16* y16  = hp;  hp += BSD;
  _Float16* att16 = hp; hp += (long)BH * S * S;
  _Float16* t116  = hp; hp += (long)BS * DFF;
  _Float16* wT    = hp;                     // L * LW halves

  const float attn_scale = 0.125f;          // hd^-0.5, hd=64

  // ---- one-time (per call) weight convert+transpose to f16 [N][K] ----
  for (int l = 0; l < L; ++l) {
    _Float16* wqT = wT + (long)l * LW;
    _Float16* wkT = wqT + (long)D * D;
    _Float16* wvT = wkT + (long)D * D;
    _Float16* woT = wvT + (long)D * D;
    _Float16* w1T = woT + (long)D * D;          // E x [DFF][D]
    _Float16* w2T = w1T + (long)E * D * DFF;    // E x [D][DFF]
    transpose_cvt<<<dim3(D / 32, D / 32), 256, 0, stream>>>(wq + (long)l * D * D, wqT, D, D);
    transpose_cvt<<<dim3(D / 32, D / 32), 256, 0, stream>>>(wk + (long)l * D * D, wkT, D, D);
    transpose_cvt<<<dim3(D / 32, D / 32), 256, 0, stream>>>(wv + (long)l * D * D, wvT, D, D);
    transpose_cvt<<<dim3(D / 32, D / 32), 256, 0, stream>>>(wo + (long)l * D * D, woT, D, D);
    for (int e = 0; e < E; ++e) {
      transpose_cvt<<<dim3(D / 32, DFF / 32), 256, 0, stream>>>(
          e_w1 + (long)(l * E + e) * D * DFF, w1T + (long)e * D * DFF, D, DFF);
      transpose_cvt<<<dim3(DFF / 32, D / 32), 256, 0, stream>>>(
          e_w2 + (long)(l * E + e) * DFF * D, w2T + (long)e * DFF * D, DFF, D);
    }
  }

  copy_k<<<dim3((unsigned)((BSD + 255) / 256)), 256, 0, stream>>>(x, x_in, BSD);

  for (int l = 0; l < L; ++l) {
    _Float16* wqT = wT + (long)l * LW;
    _Float16* wkT = wqT + (long)D * D;
    _Float16* wvT = wkT + (long)D * D;
    _Float16* woT = wvT + (long)D * D;
    _Float16* w1T = woT + (long)D * D;
    _Float16* w2T = w1T + (long)E * D * DFF;

    // ---- attention ----
    layernorm_k<<<BS, 256, 0, stream>>>(x, ln1_w + l * D, ln1_b + l * D, h32, h16, D);

    wmma_gemm<<<dim3(BS / BM, D / BN, 1), 256, 0, stream>>>(
        h16, 0, 0, D, wqT, 0, 0, D, bq + l * D,
        q16, 0, 0, D, BS, D, D, 1, F_BIAS | F_C16, 1.f, nullptr, 1);
    wmma_gemm<<<dim3(BS / BM, D / BN, 1), 256, 0, stream>>>(
        h16, 0, 0, D, wkT, 0, 0, D, bk + l * D,
        k16, 0, 0, D, BS, D, D, 1, F_BIAS | F_C16, 1.f, nullptr, 1);
    wmma_gemm<<<dim3(BS / BM, D / BN, 1), 256, 0, stream>>>(
        h16, 0, 0, D, wvT, 0, 0, D, bv + l * D,
        v16, 0, 0, D, BS, D, D, 1, F_BIAS | F_C16, 1.f, nullptr, 1);

    // scores = scale * Q @ K^T (+causal), batched over (b,h); K is already [n][k]
    wmma_gemm<<<dim3(S / BM, S / BN, BH), 256, 0, stream>>>(
        q16, (long)S * D, hd, D, k16, (long)S * D, hd, D, nullptr,
        att16, (long)H * S * S, (long)S * S, S,
        S, S, hd, H, F_CAUSAL | F_C16, attn_scale, nullptr, 1);

    softmax_rows16<<<(BH * S) / 8, 256, 0, stream>>>(att16, S);

    // vT[b,h][n][s] for the PV GEMM's [N][K] B-layout
    transpose_head16<<<dim3(S / 32, hd / 32, BH), 256, 0, stream>>>(v16, vT16, S, D, hd, H);

    // y = P @ V  (heads re-interleaved into [B,S,D])
    wmma_gemm<<<dim3(S / BM, hd / BN, BH), 256, 0, stream>>>(
        att16, (long)H * S * S, (long)S * S, S, vT16, (long)H * hd * S, (long)hd * S, S,
        nullptr, y16, (long)S * D, hd, D, S, hd, S, H, F_C16, 1.f, nullptr, 1);

    // x += y @ Wo + bo
    wmma_gemm<<<dim3(BS / BM, D / BN, 1), 256, 0, stream>>>(
        y16, 0, 0, D, woT, 0, 0, D, bo + l * D,
        x, 0, 0, D, BS, D, D, 1, F_BIAS | F_ACCUM, 1.f, nullptr, 1);

    // ---- MoE ----
    layernorm_k<<<BS, 256, 0, stream>>>(x, ln2_w + l * D, ln2_b + l * D, h32, h16, D);

    const float* rsbase; int rsdiv; unsigned skip;
    if ((l & 1) == 0) {  // token-routed
      gate_top2<<<(BS + 7) / 8, 256, 0, stream>>>(
          h32, gate_w + (long)l * D * E, gate_b + l * E, cw, BS, D);
      rsbase = cw; rsdiv = 1; skip = 0;
    } else {             // sequence-routed: per-sample weights, skippable
      seq_mean<<<(B * D + 255) / 256, 256, 0, stream>>>(h32, hmean, B, S, D);
      gate_top2<<<1, 256, 0, stream>>>(
          hmean, gate_w + (long)l * D * E, gate_b + l * E, cw2, B, D);
      rsbase = cw2; rsdiv = S; skip = F_SKIPZERO;
    }

    zero_k<<<dim3((unsigned)((BSD + 255) / 256)), 256, 0, stream>>>(macc, BSD);

    for (int e = 0; e < E; ++e) {
      const _Float16* w1 = w1T + (long)e * D * DFF;   // [DFF][D]
      const _Float16* w2 = w2T + (long)e * DFF * D;   // [D][DFF]
      const float* b1 = e_b1 + (long)(l * E + e) * DFF;
      const float* b2 = e_b2 + (long)(l * E + e) * D;
      // t1 = gelu(h @ w1 + b1)  (f16 out)
      wmma_gemm<<<dim3(BS / BM, DFF / BN, 1), 256, 0, stream>>>(
          h16, 0, 0, D, w1, 0, 0, D, b1,
          t116, 0, 0, DFF, BS, DFF, D, 1,
          F_BIAS | F_GELU | F_C16 | skip, 1.f, rsbase + e, rsdiv);
      // macc += (t1 @ w2 + b2) * cw[:, e]
      wmma_gemm<<<dim3(BS / BM, D / BN, 1), 256, 0, stream>>>(
          t116, 0, 0, DFF, w2, 0, 0, DFF, b2,
          macc, 0, 0, D, BS, D, DFF, 1,
          F_BIAS | F_ACCUM | F_ROWSCALE | skip, 1.f, rsbase + e, rsdiv);
    }

    add_k<<<dim3((unsigned)((BSD + 255) / 256)), 256, 0, stream>>>(x, macc, BSD);
  }
}
